// PointTransformerEncoder_38800734552439
// MI455X (gfx1250) — matrix operand order repaired and checked by reading, not compiled, not run on hardware
//
#include <hip/hip_runtime.h>
#include <hip/hip_bf16.h>

// ============================================================================
// Point Transformer encoder forward for gfx1250 (MI455X).
// 1x1 convs -> v_wmma_f32_16x16x32_f16 with 4-way N-tiling (A fragment reused
// across 4 back-to-back WMMAs, distinct B fragment registers per tile to avoid
// the WMMA->VALU WAR hazard NOPs), B tiles staged global->LDS with
// double-buffered global_load_async_to_lds_b128 (+ s_wait_asynccnt),
// BN/bias/ReLU/residual fused into the WMMA epilogue.
// Irregular ops (kNN, FPS, gathers, softmax, max-pool) are scalar kernels.
// ============================================================================

typedef _Float16 v16h __attribute__((ext_vector_type(16)));
typedef float    v8f  __attribute__((ext_vector_type(8)));

#define BN_EPS 1e-5f

union HFrag { v16h v; _Float16 h[16]; };

// ---------------------------------------------------------------------------
// (s,t) prep used by the non-GEMM fused consumers (pos3 / attn_pre / wsum).
// ---------------------------------------------------------------------------
__global__ void prep_affine_kernel(const float* __restrict__ gamma,
                                   const float* __restrict__ beta,
                                   const float* __restrict__ mean,
                                   const float* __restrict__ var,
                                   const float* __restrict__ bias,
                                   float* __restrict__ s, float* __restrict__ t,
                                   int C) {
  int i = blockIdx.x * blockDim.x + threadIdx.x;
  if (i >= C) return;
  float sv = 1.f, tv = 0.f;
  if (gamma) {
    sv = gamma[i] * rsqrtf(var[i] + BN_EPS);
    tv = beta[i] - mean[i] * sv;
  }
  if (bias) tv += bias[i] * sv;
  s[i] = sv; t[i] = tv;
}

// ---------------------------------------------------------------------------
// WMMA GEMM:  Y[b,m,n] = epi( sum_k W[m,k] * X[b,k,n] )
//   epi(v) = relu?( v*s[m] + t[m] (+ resid[b,m,n]) ),  s/t folded from BN+bias.
//
// One wave32 per 16 x (16*NT) output strip, NT accumulators sharing one A
// fragment per K-step; all NT B fragments are converted into distinct VGPR
// blocks before the NT WMMAs issue back-to-back (no D->A/B overlap between
// them -> no hazard NOPs).
// NT=4 (Ncols%64==0): B tile (32x64 f32) staged via double-buffered
//   global_load_async_to_lds_b128 (16 coalesced issues/tile); consume tile j
//   while tile j+1 is in flight (s_wait_asynccnt 0x10 / 0x0 on last).
// NT=1 (Ncols==16): single-buffered b32 async staging.
// K-tail (<32) uses branch-free clamped+select loads straight from global.
// Fragment layouts per CDNA5 ISA 7.12.2 (wave32). All staging/wait branches
// are wave-uniform so EXEC is all-ones at every WMMA.
// ---------------------------------------------------------------------------
template<int NT>
__global__ __launch_bounds__(32)
void gemm_wmma_kernel(const float* __restrict__ W, const float* __restrict__ X,
                      float* __restrict__ Y,
                      const float* __restrict__ gamma, const float* __restrict__ beta,
                      const float* __restrict__ mean,  const float* __restrict__ var,
                      const float* __restrict__ bias,
                      const float* __restrict__ resid,
                      int Mdim, int Kdim, int Ncols, int relu) {
  constexpr int TN = 16 * NT;                          // strip width in columns
  __shared__ float sB[(NT == 4 ? 2 : 1) * 32 * TN];    // [buf][row 0..31][col 0..TN-1]

  const int lane  = threadIdx.x;
  const int tm    = blockIdx.x * 16;
  const int tn    = blockIdx.y * TN;
  const size_t xb = (size_t)blockIdx.z * (size_t)Kdim * (size_t)Ncols;
  const size_t yb = (size_t)blockIdx.z * (size_t)Mdim * (size_t)Ncols;
  const int row   = tm + (lane & 15);          // A row for this lane
  const int scol  = lane & 15;                 // fragment column within 16-tile
  const int srow  = lane >> 4;                 // staging row parity
  const int abase = (lane >= 16) ? 8 : 0;      // A K-offset for upper half-wave
  const int bbase = (lane >= 16) ? 8 : 0;      // B K-offset for upper half-wave

  const unsigned ldsBase = (unsigned)(size_t)(void*)sB;  // LDS byte offset (addr[31:0])

  v8f acc[NT];
#pragma unroll
  for (int nt = 0; nt < NT; ++nt) acc[nt] = (v8f){0.f,0.f,0.f,0.f,0.f,0.f,0.f,0.f};
  HFrag A;
  HFrag Bq[NT];

  const int kfull = Kdim & ~31;

  if constexpr (NT == 4) {
    // ---- NT=4: b128 double-buffered async staging --------------------------
    // Tile element (16B unit) u = i*32 + lane -> row = 2*i + srow, col unit = scol.
    if (kfull > 0) {
#pragma unroll
      for (int i = 0; i < 16; ++i) {
        const float* gp = X + xb + (size_t)(2 * i + srow) * Ncols + tn + (scol << 2);
        unsigned lo = ldsBase + (unsigned)((i * 32 + lane) * 16);
        asm volatile("global_load_async_to_lds_b128 %0, %1, off"
                     :: "v"(lo), "v"(gp) : "memory");
      }
    }
    int buf = 0;
    for (int kk = 0; kk < kfull; kk += 32) {
      const int morek = (kk + 32 < kfull);
      if (morek) {                                    // prefetch next tile
#pragma unroll
        for (int i = 0; i < 16; ++i) {
          const float* gp = X + xb + (size_t)(kk + 32 + 2 * i + srow) * Ncols
                              + tn + (scol << 2);
          unsigned lo = ldsBase + (unsigned)((buf ^ 1) * (32 * TN * 4))
                                + (unsigned)((i * 32 + lane) * 16);
          asm volatile("global_load_async_to_lds_b128 %0, %1, off"
                       :: "v"(lo), "v"(gp) : "memory");
        }
      }
      // A fragment (amortized over NT WMMAs)
#pragma unroll
      for (int e = 0; e < 16; ++e) {
        int ka = kk + abase + (e < 8 ? e : 8 + e);
        A.h[e] = (_Float16)W[(size_t)row * Kdim + ka];
      }
      if (morek) asm volatile("s_wait_asynccnt 0x10" ::: "memory");  // cur tile done
      else       asm volatile("s_wait_asynccnt 0x0"  ::: "memory");
      const float* sbuf = sB + buf * (32 * TN);
      // Build all NT B fragments first (distinct VGPR blocks)...
#pragma unroll
      for (int nt = 0; nt < NT; ++nt)
#pragma unroll
        for (int e = 0; e < 16; ++e)
          Bq[nt].h[e] = (_Float16)sbuf[(bbase + (e < 8 ? e : 8 + e)) * TN + nt * 16 + scol];
      // ...then issue NT WMMAs back-to-back (no inter-WMMA hazards).
#pragma unroll
      for (int nt = 0; nt < NT; ++nt)
        acc[nt] = __builtin_amdgcn_wmma_f32_16x16x32_f16(false, A.v, false, Bq[nt].v,
                                                         (short)0, acc[nt], false, false);
      buf ^= 1;
    }
  } else {
    // ---- NT=1: b32 single-buffered async staging ---------------------------
    for (int kk = 0; kk < kfull; kk += 32) {
#pragma unroll
      for (int e = 0; e < 16; ++e) {
        int ka = kk + abase + (e < 8 ? e : 8 + e);
        A.h[e] = (_Float16)W[(size_t)row * Kdim + ka];
      }
#pragma unroll
      for (int i = 0; i < 16; ++i) {
        const float* gp = X + xb + (size_t)(kk + 2 * i + srow) * Ncols + tn + scol;
        unsigned lo = ldsBase + (unsigned)((i * 32 + lane) * 4);
        asm volatile("global_load_async_to_lds_b32 %0, %1, off"
                     :: "v"(lo), "v"(gp) : "memory");
      }
      asm volatile("s_wait_asynccnt 0x0" ::: "memory");
#pragma unroll
      for (int e = 0; e < 16; ++e)
        Bq[0].h[e] = (_Float16)sB[(bbase + (e < 8 ? e : 8 + e)) * TN + scol];
      acc[0] = __builtin_amdgcn_wmma_f32_16x16x32_f16(false, A.v, false, Bq[0].v,
                                                      (short)0, acc[0], false, false);
    }
  }

  // ---- branch-free K tail (Kdim % 32: Kdim in {3,35,67,131,259}) ----
  if (kfull < Kdim) {
    const int kk = kfull;
    const int kmax = Kdim - 1;
#pragma unroll
    for (int e = 0; e < 16; ++e) {
      int ka = kk + abase + (e < 8 ? e : 8 + e);
      int kac = ka < kmax ? ka : kmax;                       // clamped, always legal
      float wv = W[(size_t)row * Kdim + kac];
      A.h[e] = (ka < Kdim) ? (_Float16)wv : (_Float16)0.f;   // v_cndmask select
    }
#pragma unroll
    for (int nt = 0; nt < NT; ++nt) {
      const int col = tn + nt * 16 + scol;
#pragma unroll
      for (int e = 0; e < 16; ++e) {
        int kb = kk + bbase + (e < 8 ? e : 8 + e);
        int kbc = kb < kmax ? kb : kmax;
        float xv = X[xb + (size_t)kbc * Ncols + col];
        Bq[nt].h[e] = (kb < Kdim) ? (_Float16)xv : (_Float16)0.f;
      }
    }
#pragma unroll
    for (int nt = 0; nt < NT; ++nt)
      acc[nt] = __builtin_amdgcn_wmma_f32_16x16x32_f16(false, A.v, false, Bq[nt].v,
                                                       (short)0, acc[nt], false, false);
  }

  // ---- epilogue: fused BN/bias (+residual) (+relu) ----
  const int rbase = (lane >= 16) ? 8 : 0;
#pragma unroll
  for (int v = 0; v < 8; ++v) {
    int m = tm + rbase + v;
    float sv = 1.f, tv = 0.f;
    if (gamma) {
      sv = gamma[m] * rsqrtf(var[m] + BN_EPS);
      tv = beta[m] - mean[m] * sv;
    }
    if (bias) tv += bias[m] * sv;
#pragma unroll
    for (int nt = 0; nt < NT; ++nt) {
      const int col = tn + nt * 16 + scol;
      float val = acc[nt][v] * sv + tv;
      if (resid) val += resid[yb + (size_t)m * Ncols + col];
      if (relu)  val = fmaxf(val, 0.f);
      Y[yb + (size_t)m * Ncols + col] = val;
    }
  }
}

// ---------------------------------------------------------------------------
// pc [B,N,3] -> x3 [B,3,N]
// ---------------------------------------------------------------------------
__global__ void transpose_pc_kernel(const float* __restrict__ pc,
                                    float* __restrict__ x3, int N, int total) {
  int i = blockIdx.x * blockDim.x + threadIdx.x;
  if (i >= total) return;
  int b = i / N, n = i % N;
  for (int c = 0; c < 3; ++c)
    x3[(size_t)b * 3 * N + (size_t)c * N + n] = pc[((size_t)b * N + n) * 3 + c];
}

// ---------------------------------------------------------------------------
// kNN: indices of K nearest (ascending d2, ties -> lower index).
// ---------------------------------------------------------------------------
__global__ void knn_kernel(const float* __restrict__ pq, const float* __restrict__ pa,
                           int* __restrict__ idx, int M, int N, int K, int total) {
  int gid = blockIdx.x * blockDim.x + threadIdx.x;
  if (gid >= total) return;
  int b = gid / M, m = gid % M;
  const float* q = pq + ((size_t)b * M + m) * 3;
  float qx = q[0], qy = q[1], qz = q[2];
  const float* a = pa + (size_t)b * N * 3;
  float kd[16]; int ki[16];
  for (int j = 0; j < K; ++j) { kd[j] = 3.4e38f; ki[j] = 0; }
  for (int n = 0; n < N; ++n) {
    float dx = a[n * 3 + 0] - qx, dy = a[n * 3 + 1] - qy, dz = a[n * 3 + 2] - qz;
    float d2 = dx * dx + dy * dy + dz * dz;
    if (d2 < kd[K - 1]) {
      int j = K - 1;
      while (j > 0 && d2 < kd[j - 1]) { kd[j] = kd[j - 1]; ki[j] = ki[j - 1]; --j; }
      kd[j] = d2; ki[j] = n;
    }
  }
  int* o = idx + ((size_t)b * M + m) * K;
  for (int j = 0; j < K; ++j) o[j] = ki[j];
}

// ---------------------------------------------------------------------------
// Farthest point sampling (start idx 0). One workgroup per batch; distances
// in LDS; argmax tie-break = first (lowest) index, matching jnp.argmax.
// ---------------------------------------------------------------------------
__global__ __launch_bounds__(256)
void fps_kernel(const float* __restrict__ p, int* __restrict__ fidx, int N, int M) {
  __shared__ float d[4096];
  __shared__ float rv[256];
  __shared__ int   ri[256];
  __shared__ int   cur;
  int b = blockIdx.x, tid = threadIdx.x;
  const float* pb = p + (size_t)b * N * 3;
  float x0 = pb[0], y0 = pb[1], z0 = pb[2];
  for (int n = tid; n < N; n += 256) {
    float dx = pb[n * 3] - x0, dy = pb[n * 3 + 1] - y0, dz = pb[n * 3 + 2] - z0;
    d[n] = dx * dx + dy * dy + dz * dz;
  }
  if (tid == 0) fidx[(size_t)b * M] = 0;
  __syncthreads();
  for (int j = 1; j < M; ++j) {
    float bv = -1.f; int bi = 0x7fffffff;
    for (int n = tid; n < N; n += 256) {
      float v = d[n];
      if (v > bv || (v == bv && n < bi)) { bv = v; bi = n; }
    }
    rv[tid] = bv; ri[tid] = bi;
    __syncthreads();
    for (int sft = 128; sft > 0; sft >>= 1) {
      if (tid < sft) {
        float v2 = rv[tid + sft]; int i2 = ri[tid + sft];
        if (v2 > rv[tid] || (v2 == rv[tid] && i2 < ri[tid])) { rv[tid] = v2; ri[tid] = i2; }
      }
      __syncthreads();
    }
    if (tid == 0) { cur = ri[0]; fidx[(size_t)b * M + j] = cur; }
    __syncthreads();
    int i = cur;
    float cx = pb[i * 3], cy = pb[i * 3 + 1], cz = pb[i * 3 + 2];
    for (int n = tid; n < N; n += 256) {
      float dx = pb[n * 3] - cx, dy = pb[n * 3 + 1] - cy, dz = pb[n * 3 + 2] - cz;
      float nd = dx * dx + dy * dy + dz * dz;
      if (nd < d[n]) d[n] = nd;
    }
    __syncthreads();
  }
}

// p2[b,m,:] = p1[b, fidx[b,m], :]
__global__ void gather_pts_kernel(const float* __restrict__ p1, const int* __restrict__ fidx,
                                  float* __restrict__ p2, int M, int N, int total) {
  int gid = blockIdx.x * blockDim.x + threadIdx.x;
  if (gid >= total) return;
  int b = gid / M, m = gid % M;
  int i = fidx[(size_t)b * M + m];
  for (int c = 0; c < 3; ++c)
    p2[((size_t)b * M + m) * 3 + c] = p1[((size_t)b * N + i) * 3 + c];
}

// rel[b,c,m,k] = pa[b, idx[b,m,k], c] - pq[b,m,c]     (rel is [B,3,M,K])
__global__ void build_rel_kernel(const float* __restrict__ pq, const float* __restrict__ pa,
                                 const int* __restrict__ idx, float* __restrict__ rel,
                                 int M, int N, int K, int total) {
  int gid = blockIdx.x * blockDim.x + threadIdx.x;
  if (gid >= total) return;
  int b = gid / (M * K), r = gid % (M * K), m = r / K, k = r % K;
  int src = idx[gid];
  for (int c = 0; c < 3; ++c)
    rel[(((size_t)b * 3 + c) * M + m) * K + k] =
        pa[((size_t)b * N + src) * 3 + c] - pq[((size_t)b * M + m) * 3 + c];
}

// g[b,c,m,k] = f[b,c,idx[b,m,k]]
__global__ void gather_feat_kernel(const float* __restrict__ f, const int* __restrict__ idx,
                                   float* __restrict__ g, int C, int N, int M, int K,
                                   int total) {
  int gid = blockIdx.x * blockDim.x + threadIdx.x;
  if (gid >= total) return;
  int b = gid / (M * K), r = gid % (M * K), m = r / K, k = r % K;
  int src = idx[gid];
  for (int c = 0; c < C; ++c)
    g[(((size_t)b * C + c) * M + m) * K + k] = f[((size_t)b * C + c) * N + src];
}

// TD gather+concat: g [B,3+C,M,K] = [rel ; gathered feat]
__global__ void td_gather_kernel(const float* __restrict__ pq, const float* __restrict__ pa,
                                 const int* __restrict__ idx, const float* __restrict__ f,
                                 float* __restrict__ g, int C, int N, int M, int K,
                                 int total) {
  int gid = blockIdx.x * blockDim.x + threadIdx.x;
  if (gid >= total) return;
  int b = gid / (M * K), r = gid % (M * K), m = r / K, k = r % K;
  int src = idx[gid];
  int Ct = 3 + C;
  for (int c = 0; c < 3; ++c)
    g[(((size_t)b * Ct + c) * M + m) * K + k] =
        pa[((size_t)b * N + src) * 3 + c] - pq[((size_t)b * M + m) * 3 + c];
  for (int c = 0; c < C; ++c)
    g[(((size_t)b * Ct + 3 + c) * M + m) * K + k] = f[((size_t)b * C + c) * N + src];
}

// r = relu(bn(pos_w1 @ rel)) on [B,3,R]  (3x3 conv, too small for WMMA)
__global__ void pos3_kernel(const float* __restrict__ w, const float* __restrict__ rel,
                            const float* __restrict__ s, const float* __restrict__ t,
                            float* __restrict__ r, int R, int total) {
  int gid = blockIdx.x * blockDim.x + threadIdx.x;
  if (gid >= total) return;
  int b = gid / R, rr = gid % R;
  size_t base = (size_t)b * 3 * R + rr;
  float i0 = rel[base], i1 = rel[base + R], i2 = rel[base + 2 * (size_t)R];
  for (int o = 0; o < 3; ++o) {
    float acc = w[o * 3] * i0 + w[o * 3 + 1] * i1 + w[o * 3 + 2] * i2;
    r[(size_t)b * 3 * R + (size_t)o * R + rr] = fmaxf(s[o] * acc + t[o], 0.f);
  }
}

// a1 = relu(bn1(q[...,None] - n_k + n_r)) on [B,C,N,K]
__global__ void attn_pre_kernel(const float* __restrict__ q, const float* __restrict__ nk,
                                const float* __restrict__ nr, const float* __restrict__ s,
                                const float* __restrict__ t, float* __restrict__ a1,
                                int C, int N, int K, int total) {
  int gid = blockIdx.x * blockDim.x + threadIdx.x;
  if (gid >= total) return;
  int NK = N * K;
  int b = gid / (C * NK), rem = gid % (C * NK), c = rem / NK, r2 = rem % NK, n = r2 / K;
  float v = q[((size_t)b * C + c) * N + n] - nk[gid] + nr[gid];
  a1[gid] = fmaxf(s[c] * v + t[c], 0.f);
}

// softmax over K (in place), rows = B*C*N
__global__ void softmax_kernel(float* __restrict__ a, int K, int rows) {
  int gid = blockIdx.x * blockDim.x + threadIdx.x;
  if (gid >= rows) return;
  size_t base = (size_t)gid * K;
  float mx = a[base];
  for (int k = 1; k < K; ++k) mx = fmaxf(mx, a[base + k]);
  float sum = 0.f;
  for (int k = 0; k < K; ++k) { float e = expf(a[base + k] - mx); a[base + k] = e; sum += e; }
  float inv = 1.f / sum;
  for (int k = 0; k < K; ++k) a[base + k] *= inv;
}

// out[b,c,n] = relu(s[c]*sum_k (n_v+n_r)*a + t[c]), rows = B*C*N
__global__ void wsum_kernel(const float* __restrict__ nv, const float* __restrict__ nr,
                            const float* __restrict__ a, const float* __restrict__ s,
                            const float* __restrict__ t, float* __restrict__ out,
                            int C, int N, int K, int rows) {
  int gid = blockIdx.x * blockDim.x + threadIdx.x;
  if (gid >= rows) return;
  int c = (gid / N) % C;
  size_t base = (size_t)gid * K;
  float acc = 0.f;
  for (int k = 0; k < K; ++k) acc += (nv[base + k] + nr[base + k]) * a[base + k];
  out[gid] = fmaxf(s[c] * acc + t[c], 0.f);
}

// out[b,c,m] = max_k in[b,c,m,k], rows = B*C*M
__global__ void maxk_kernel(const float* __restrict__ in, float* __restrict__ out,
                            int K, int rows) {
  int gid = blockIdx.x * blockDim.x + threadIdx.x;
  if (gid >= rows) return;
  size_t base = (size_t)gid * K;
  float mx = in[base];
  for (int k = 1; k < K; ++k) mx = fmaxf(mx, in[base + k]);
  out[gid] = mx;
}

// ============================================================================
// Host side
// ============================================================================

namespace {

struct BNP { const float *beta, *gamma, *mean, *var; };
struct PTLP {
  const float* attn_b2; BNP attn_bn1, attn_bn2;
  const float *attn_w1, *attn_w2, *k_b, *k_w, *pos_b2;
  BNP pos_bn; const float *pos_w1, *pos_w2, *q_b, *q_w, *v_b, *v_w;
};
struct BlockP { BNP bn, bn1, bn2; PTLP layer; const float *lin1_w, *lin2_w; };
struct TDP { BNP bn1, bn2; const float *w1, *w2; };

struct Cursor {
  void* const* din; int i;
  const float* next() { return (const float*)din[i++]; }
};
static BNP readBN(Cursor& c) {
  BNP b; b.beta = c.next(); b.gamma = c.next(); b.mean = c.next(); b.var = c.next();
  return b;
}
static PTLP readPTL(Cursor& c) {
  PTLP p;
  p.attn_b2 = c.next(); p.attn_bn1 = readBN(c); p.attn_bn2 = readBN(c);
  p.attn_w1 = c.next(); p.attn_w2 = c.next();
  p.k_b = c.next(); p.k_w = c.next();
  p.pos_b2 = c.next(); p.pos_bn = readBN(c); p.pos_w1 = c.next(); p.pos_w2 = c.next();
  p.q_b = c.next(); p.q_w = c.next(); p.v_b = c.next(); p.v_w = c.next();
  return p;
}
static BlockP readBlock(Cursor& c) {
  BlockP b;
  b.bn = readBN(c); b.bn1 = readBN(c); b.bn2 = readBN(c);
  b.layer = readPTL(c); b.lin1_w = c.next(); b.lin2_w = c.next();
  return b;
}
static TDP readTD(Cursor& c) {
  TDP t; t.bn1 = readBN(c); t.bn2 = readBN(c); t.w1 = c.next(); t.w2 = c.next();
  return t;
}

static inline int cdiv(int a, int b) { return (a + b - 1) / b; }

// WMMA GEMM with fully fused epilogue (BN/bias/residual/relu); picks the
// 4-way N-tiled b128 double-buffered kernel when columns allow.
static void conv_gemm(hipStream_t st, const float* W, const float* X, float* Y,
                      const BNP* bn, const float* bias, const float* resid, int relu,
                      int Mdim, int Kdim, int Ncols, int B) {
  const float* g = bn ? bn->gamma : nullptr;
  const float* be = bn ? bn->beta : nullptr;
  const float* mn = bn ? bn->mean : nullptr;
  const float* vr = bn ? bn->var : nullptr;
  if ((Ncols & 63) == 0) {
    dim3 gr(Mdim / 16, Ncols / 64, B);
    gemm_wmma_kernel<4><<<gr, 32, 0, st>>>(W, X, Y, g, be, mn, vr, bias, resid,
                                           Mdim, Kdim, Ncols, relu);
  } else {
    dim3 gr(Mdim / 16, Ncols / 16, B);
    gemm_wmma_kernel<1><<<gr, 32, 0, st>>>(W, X, Y, g, be, mn, vr, bias, resid,
                                           Mdim, Kdim, Ncols, relu);
  }
}

} // namespace

extern "C" void kernel_launch(void* const* d_in, const int* in_sizes, int n_in,
                              void* d_out, int out_size, void* d_ws, size_t ws_size,
                              hipStream_t stream) {
  (void)in_sizes; (void)n_in; (void)out_size; (void)ws_size;

  const int B = 2;
  static const int PLANES[5]  = {32, 64, 128, 256, 512};
  static const int BLOCKS[5]  = {2, 3, 4, 6, 3};
  static const int NSAMP[5]   = {8, 16, 16, 16, 16};

  const float* pc = (const float*)d_in[0];
  // d_in[1] = feat (unused by the reference forward)

  // ---- walk the params pytree (jax tree-flatten order: sorted dict keys) ----
  Cursor cur{d_in, 2};
  BNP dec_bn1 = readBN(cur);
  BNP dec_bn2 = readBN(cur);
  const float* dec_w1 = cur.next();
  const float* dec_w2 = cur.next();
  BNP in_bn1 = readBN(cur);
  BNP in_bn2 = readBN(cur);
  const float* in_w1 = cur.next();
  const float* in_w2 = cur.next();
  BlockP blocks[5][6];
  TDP    tds[5];
  for (int si = 0; si < 5; ++si) {
    for (int bi = 0; bi < BLOCKS[si]; ++bi) blocks[si][bi] = readBlock(cur);
    if (si > 0) tds[si] = readTD(cur);   // stage 0 td == None (dropped leaf)
  }

  // ---- workspace carve (floats) ----
  const size_t ACT_CAP  = (size_t)B * 32 * 4096;        // 262144
  const size_t P_CAP    = (size_t)B * 4096 * 3;         // 24576
  const size_t REL_CAP  = (size_t)B * 3 * 4096 * 8;     // 196608
  const size_t D4_CAP   = (size_t)B * 64 * 1024 * 16;   // 2097152 (== B*32*4096*8)
  const size_t IDX_CAP  = (size_t)B * 4096 * 8;         // 65536
  const size_t FIDX_CAP = (size_t)B * 1024;             // 2048

  float* wsf = (float*)d_ws;
  size_t off = 0;
  auto alloc = [&](size_t n) { float* p = wsf + off; off += n; return p; };
  float* xA   = alloc(ACT_CAP);
  float* xB   = alloc(ACT_CAP);
  float* y1b  = alloc(ACT_CAP);
  float* y2b  = alloc(ACT_CAP);
  float* qb   = alloc(ACT_CAP);
  float* kb   = alloc(ACT_CAP);
  float* vb   = alloc(ACT_CAP);
  float* pA   = alloc(P_CAP);
  float* pB   = alloc(P_CAP);
  float* relb = alloc(REL_CAP);
  float* rb   = alloc(REL_CAP);
  float* nk   = alloc(D4_CAP);
  float* nv   = alloc(D4_CAP);
  float* nr   = alloc(D4_CAP);
  float* a1   = alloc(D4_CAP);
  float* a2   = alloc(D4_CAP);
  float* a3   = alloc(D4_CAP);
  float* sb   = alloc(512);
  float* tb   = alloc(512);
  int* idxb  = (int*)alloc(IDX_CAP);
  int* fidxb = (int*)alloc(FIDX_CAP);

  const int T = 256;

  // ---- input MLP: x = relu(bn2(W2 @ relu(bn1(W1 @ pc^T)))) ----
  int N = 4096, C = 32;
  transpose_pc_kernel<<<cdiv(B * N, T), T, 0, stream>>>(pc, y1b, N, B * N);
  conv_gemm(stream, in_w1, y1b, xA, &in_bn1, nullptr, nullptr, 1, 32, 3, N, B);
  conv_gemm(stream, in_w2, xA, xB, &in_bn2, nullptr, nullptr, 1, 32, 32, N, B);
  float* x  = xB;
  float* xo = xA;
  const float* pcur = pc;

  for (int si = 0; si < 5; ++si) {
    const int k = NSAMP[si];

    if (si > 0) {
      // ---------- transition down ----------
      const int Cout = PLANES[si];
      const int M = N / 4;
      float* pn = (pcur == pA) ? pB : pA;
      fps_kernel<<<B, 256, 0, stream>>>(pcur, fidxb, N, M);
      gather_pts_kernel<<<cdiv(B * M, T), T, 0, stream>>>(pcur, fidxb, pn, M, N, B * M);
      knn_kernel<<<cdiv(B * M, T), T, 0, stream>>>(pn, pcur, idxb, M, N, 16, B * M);
      td_gather_kernel<<<cdiv(B * M * 16, T), T, 0, stream>>>(
          pn, pcur, idxb, x, a1, C, N, M, 16, B * M * 16);
      conv_gemm(stream, tds[si].w1, a1, a2, &tds[si].bn1, nullptr, nullptr, 1,
                Cout, 3 + C, M * 16, B);
      conv_gemm(stream, tds[si].w2, a2, a3, &tds[si].bn2, nullptr, nullptr, 1,
                Cout, Cout, M * 16, B);
      maxk_kernel<<<cdiv(B * Cout * M, T), T, 0, stream>>>(a3, xo, 16, B * Cout * M);
      { float* tmp = x; x = xo; xo = tmp; }
      pcur = pn; N = M; C = Cout;
    }

    for (int bi = 0; bi < BLOCKS[si]; ++bi) {
      const BlockP& bp = blocks[si][bi];
      const PTLP& L = bp.layer;
      const int NK = N * k;

      // y = relu(bn1(lin1 @ x))
      conv_gemm(stream, bp.lin1_w, x, y1b, &bp.bn1, nullptr, nullptr, 1, C, C, N, B);

      // q/k/v projections (bias only)
      conv_gemm(stream, L.q_w, y1b, qb, nullptr, L.q_b, nullptr, 0, C, C, N, B);
      conv_gemm(stream, L.k_w, y1b, kb, nullptr, L.k_b, nullptr, 0, C, C, N, B);
      conv_gemm(stream, L.v_w, y1b, vb, nullptr, L.v_b, nullptr, 0, C, C, N, B);

      // neighborhoods
      knn_kernel<<<cdiv(B * N, T), T, 0, stream>>>(pcur, pcur, idxb, N, N, k, B * N);
      gather_feat_kernel<<<cdiv(B * NK, T), T, 0, stream>>>(kb, idxb, nk, C, N, N, k, B * NK);
      gather_feat_kernel<<<cdiv(B * NK, T), T, 0, stream>>>(vb, idxb, nv, C, N, N, k, B * NK);
      build_rel_kernel<<<cdiv(B * NK, T), T, 0, stream>>>(pcur, pcur, idxb, relb, N, N, k, B * NK);

      // positional MLP: r = relu(pos_bn(pos_w1 @ rel)); n_r = pos_w2 @ r + pos_b2
      prep_affine_kernel<<<1, 256, 0, stream>>>(L.pos_bn.gamma, L.pos_bn.beta,
                                                L.pos_bn.mean, L.pos_bn.var,
                                                nullptr, sb, tb, 3);
      pos3_kernel<<<cdiv(B * NK, T), T, 0, stream>>>(L.pos_w1, relb, sb, tb, rb, NK, B * NK);
      conv_gemm(stream, L.pos_w2, rb, nr, nullptr, L.pos_b2, nullptr, 0, C, 3, NK, B);

      // attention MLP
      prep_affine_kernel<<<cdiv(C, 256), 256, 0, stream>>>(L.attn_bn1.gamma, L.attn_bn1.beta,
                                                           L.attn_bn1.mean, L.attn_bn1.var,
                                                           nullptr, sb, tb, C);
      attn_pre_kernel<<<cdiv(B * C * NK, T), T, 0, stream>>>(qb, nk, nr, sb, tb, a1,
                                                             C, N, k, B * C * NK);
      conv_gemm(stream, L.attn_w1, a1, a2, &L.attn_bn2, nullptr, nullptr, 1, C, C, NK, B);
      conv_gemm(stream, L.attn_w2, a2, a3, nullptr, L.attn_b2, nullptr, 0, C, C, NK, B);
      softmax_kernel<<<cdiv(B * C * N, T), T, 0, stream>>>(a3, k, B * C * N);

      // z = relu(bn(sum_k (n_v+n_r) * softmax(a)))
      prep_affine_kernel<<<cdiv(C, 256), 256, 0, stream>>>(bp.bn.gamma, bp.bn.beta,
                                                           bp.bn.mean, bp.bn.var,
                                                           nullptr, sb, tb, C);
      wsum_kernel<<<cdiv(B * C * N, T), T, 0, stream>>>(nv, nr, a3, sb, tb, y2b,
                                                        C, N, k, B * C * N);

      // x = relu(bn2(lin2 @ z) + x)
      conv_gemm(stream, bp.lin2_w, y2b, xo, &bp.bn2, nullptr, x, 1, C, C, N, B);
      { float* tmp = x; x = xo; xo = tmp; }
    }
  }

  // ---- decoder: two conv+bn+relu at C=512, N=16; last writes d_out ----
  conv_gemm(stream, dec_w1, x, y1b, &dec_bn1, nullptr, nullptr, 1, 512, 512, N, B);
  conv_gemm(stream, dec_w2, y1b, (float*)d_out, &dec_bn2, nullptr, nullptr, 1,
            512, 512, N, B);
}
